// RPN_31997506355980
// MI455X (gfx1250) — compile-verified, matrix operand order checked by reading, not run on hardware
//
#include <hip/hip_runtime.h>

typedef __attribute__((ext_vector_type(16))) __bf16 bf16x16;
typedef __attribute__((ext_vector_type(8)))  float  f32x8;
typedef int v4i __attribute__((ext_vector_type(4)));

#define GW 100
#define GH 100
#define NPOS 10000
#define NCH 512
#define NANCH 90000
#define KTOT 4608
#define IMG 1600.0f
#define NKEYS 131072
#define NTOP 10000
#define NWORDS 157
#define NOUT 2000
#define NBLK 352

#if __has_builtin(__builtin_amdgcn_global_load_async_to_lds_b128) && \
    __has_builtin(__builtin_amdgcn_s_wait_asynccnt)
#define HAS_ASYNC 1
#else
#define HAS_ASYNC 0
#endif

// 16-byte global -> LDS copy. Async (CDNA5 GLOBAL_LOAD_ASYNC_TO_LDS_B128) when
// the toolchain exposes it, otherwise a VGPR round-trip.
__device__ __forceinline__ void cp16(const unsigned short* g, unsigned short* l) {
#if HAS_ASYNC
  __builtin_amdgcn_global_load_async_to_lds_b128(
      (__attribute__((address_space(1))) v4i*)(v4i*)(void*)(unsigned short*)g,
      (__attribute__((address_space(3))) v4i*)(v4i*)(void*)l, 0, 0);
#else
  *(uint4*)l = *(const uint4*)g;
#endif
}
__device__ __forceinline__ void cpwait3() {
#if HAS_ASYNC
  __builtin_amdgcn_s_wait_asynccnt(3);
#endif
}
__device__ __forceinline__ void cpwait0() {
#if HAS_ASYNC
  __builtin_amdgcn_s_wait_asynccnt(0);
#endif
}

__device__ __forceinline__ unsigned short f2bf(float x) {
  unsigned u = __float_as_uint(x);
  unsigned r = (u + 0x7FFFu + ((u >> 16) & 1u)) >> 16;
  return (unsigned short)r;
}

__device__ __forceinline__ float iou_ga(float gx0, float gy0, float gx1, float gy1,
                                        float ax0, float ay0, float ax1, float ay1) {
  float areaA = (gx1 - gx0) * (gy1 - gy0);
  float areaB = (ax1 - ax0) * (ay1 - ay0);
  float lx = fmaxf(gx0, ax0), ly = fmaxf(gy0, ay0);
  float rx = fminf(gx1, ax1), ry = fminf(gy1, ay1);
  float w = fmaxf(rx - lx, 0.f), h = fmaxf(ry - ly, 0.f);
  float inter = w * h;
  return inter / (areaA + areaB - inter);
}

// ---------------- init ----------------
__global__ void k_init_out(float* out, int n) {
  int i = blockIdx.x * 256 + threadIdx.x;
  if (i < n) out[i] = 0.f;
}

// ---------------- prep: feat NCHW f32 -> padded NHWC bf16 [102][102][512] ----------------
__global__ void k_featpad(const float* __restrict__ feat, unsigned short* __restrict__ fp) {
  int i = blockIdx.x * 256 + threadIdx.x;
  const int total = 102 * 102 * 512;
  if (i >= total) return;
  int cin = i & 511;
  int sp = i >> 9;
  int x = sp % 102, y = sp / 102;
  float v = 0.f;
  if (x >= 1 && x <= 100 && y >= 1 && y <= 100)
    v = feat[cin * (GW * GH) + (y - 1) * GW + (x - 1)];
  fp[i] = f2bf(v);
}

// ---------------- prep: rpn_w [O][I][3][3] -> Bw [O=512][K=4608] bf16, K=(ky*3+kx)*512+cin ----------------
__global__ void k_wconv(const float* __restrict__ w, unsigned short* __restrict__ bw) {
  int i = blockIdx.x * 256 + threadIdx.x;
  const int total = 512 * KTOT;
  if (i >= total) return;
  int o = i / KTOT;
  int k = i - o * KTOT;
  int s = k >> 9;        // ky*3+kx
  int cin = k & 511;
  bw[i] = f2bf(w[(o * 512 + cin) * 9 + s]);
}

// ---------------- prep: cls_w (9x512) + bbox_w (36x512) -> B2 [48][512] bf16 (rows 45..47 zero) ----------------
__global__ void k_w2(const float* __restrict__ cw, const float* __restrict__ bwt,
                     unsigned short* __restrict__ b2) {
  int i = blockIdx.x * 256 + threadIdx.x;
  const int total = 48 * 512;
  if (i >= total) return;
  int n = i >> 9, k = i & 511;
  float v = 0.f;
  if (n < 9) v = cw[n * 512 + k];
  else if (n < 45) v = bwt[(n - 9) * 512 + k];
  b2[i] = f2bf(v);
}

// ---------------- GEMM1: implicit-GEMM 3x3 conv + bias + relu -> rpn_feat bf16 [10000][512] ----------------
union Frag { bf16x16 f; uint4 q[2]; };

__global__ __launch_bounds__(256) void k_gemm1(const unsigned short* __restrict__ fp,
                                               const unsigned short* __restrict__ bw,
                                               const float* __restrict__ bias,
                                               unsigned short* __restrict__ rf) {
  const int LDT = 40;
  const int NT = KTOT / 32;  // 144 K-steps
  __shared__ __align__(16) unsigned short sA[2][64 * LDT];
  __shared__ __align__(16) unsigned short sB[2][128 * LDT];
  int tid = threadIdx.x;
  int wv = tid >> 5, lane = tid & 31;
  int bm = blockIdx.x * 64;
  int bn = blockIdx.y * 128;
  int waveM = (wv >> 2) * 32;
  int waveN = (wv & 3) * 32;
  f32x8 acc[2][2] = {};

  int arow = tid >> 2;          // 0..63
  int acol = (tid & 3) * 8;     // 0,8,16,24
  int p = bm + arow;
  bool pOK = p < NPOS;
  int r = pOK ? (p / GW) : 0;
  int c = pOK ? (p - r * GW) : 0;

  int brow = tid >> 1;          // 0..127
  int bcol = (tid & 1) * 16;
  int o = bn + brow;

  int lrow = lane & 15;
  int khalf = (lane < 16) ? 0 : 8;

  // stage K-tile t into LDS buffer `bs`
  auto issue = [&](int t, int bs) {
    int kt = t * 32;
    int s = kt >> 9;
    int c0 = kt & 511;
    int ky = s / 3, kx = s - ky * 3;
    // rows past M use the zeroed padded border (y==0 row of fp is all zeros)
    const unsigned short* asrc =
        pOK ? (fp + (((r + ky) * 102 + (c + kx)) * 512 + c0 + acol)) : (fp + acol);
    cp16(asrc, &sA[bs][arow * LDT + acol]);
    const unsigned short* bsrc = bw + (size_t)o * KTOT + kt + bcol;
    cp16(bsrc, &sB[bs][brow * LDT + bcol]);
    cp16(bsrc + 8, &sB[bs][brow * LDT + bcol + 8]);
  };

  auto compute = [&](int bs) {
    Frag aF[2], bF[2];
#pragma unroll
    for (int i = 0; i < 2; i++) {
      int rowi = waveM + i * 16 + lrow;
      aF[i].q[0] = *(const uint4*)&sA[bs][rowi * LDT + khalf];
      aF[i].q[1] = *(const uint4*)&sA[bs][rowi * LDT + 16 + khalf];
    }
#pragma unroll
    for (int j = 0; j < 2; j++) {
      int rowj = waveN + j * 16 + lrow;
      bF[j].q[0] = *(const uint4*)&sB[bs][rowj * LDT + khalf];
      bF[j].q[1] = *(const uint4*)&sB[bs][rowj * LDT + 16 + khalf];
    }
#pragma unroll
    for (int i = 0; i < 2; i++)
#pragma unroll
      for (int j = 0; j < 2; j++)
        acc[i][j] = __builtin_amdgcn_wmma_f32_16x16x32_bf16(
            false, aF[i].f, false, bF[j].f, (short)0, acc[i][j], false, false);
  };

  issue(0, 0);
  for (int it = 0; it < NT - 1; ++it) {
    int cur = it & 1;
    issue(it + 1, cur ^ 1);   // prefetch next tile into the other buffer
    cpwait3();                // tile `it` (3 older async ops) has landed
    __syncthreads();
    compute(cur);
    __syncthreads();
  }
  cpwait0();
  __syncthreads();
  compute((NT - 1) & 1);

  int col = lane & 15;
#pragma unroll
  for (int i = 0; i < 2; i++)
#pragma unroll
    for (int j = 0; j < 2; j++)
#pragma unroll
      for (int v = 0; v < 8; v++) {
        int row = (lane < 16) ? v : (8 + v);
        int p2 = bm + waveM + i * 16 + row;
        int o2 = bn + waveN + j * 16 + col;
        if (p2 < NPOS) {
          float x = acc[i][j][v] + bias[o2];
          x = fmaxf(x, 0.f);
          rf[(size_t)p2 * 512 + o2] = f2bf(x);
        }
      }
}

// ---------------- GEMM2: rpn_feat [10000][512] x B2 [48][512]^T -> logits / box_pred ----------------
__global__ __launch_bounds__(256) void k_gemm2(const unsigned short* __restrict__ rf,
                                               const unsigned short* __restrict__ b2,
                                               const float* __restrict__ cls_b,
                                               const float* __restrict__ bbox_b,
                                               float* __restrict__ logits,
                                               float* __restrict__ box_pred) {
  const int LDT = 40;
  __shared__ __align__(16) unsigned short sA[128 * LDT];
  __shared__ __align__(16) unsigned short sB[48 * LDT];
  int tid = threadIdx.x;
  int wv = tid >> 5, lane = tid & 31;
  int bm = blockIdx.x * 128;
  f32x8 acc[3] = {};

  int arow = tid >> 1;
  int acolh = (tid & 1) * 16;
  int p = bm + arow;
  bool pOK = p < NPOS;
  const unsigned short* zsrc = b2 + 45 * 512;  // rows 45..47 of B2 are zero

  int lrow = lane & 15;
  int khalf = (lane < 16) ? 0 : 8;

  for (int kt = 0; kt < 512; kt += 32) {
    const unsigned short* src = pOK ? (rf + (size_t)p * 512 + kt + acolh) : (zsrc + acolh);
    cp16(src, &sA[arow * LDT + acolh]);
    cp16(src + 8, &sA[arow * LDT + acolh + 8]);
    if (tid < 192) {  // waves 0..5, uniform per wave
      int rrow = tid >> 2;
      int rcol = (tid & 3) * 8;
      cp16(b2 + rrow * 512 + kt + rcol, &sB[rrow * LDT + rcol]);
    }
    cpwait0();
    __syncthreads();
    Frag aF, bF[3];
    {
      int rowi = wv * 16 + lrow;
      aF.q[0] = *(const uint4*)&sA[rowi * LDT + khalf];
      aF.q[1] = *(const uint4*)&sA[rowi * LDT + 16 + khalf];
    }
#pragma unroll
    for (int j = 0; j < 3; j++) {
      int rowj = j * 16 + lrow;
      bF[j].q[0] = *(const uint4*)&sB[rowj * LDT + khalf];
      bF[j].q[1] = *(const uint4*)&sB[rowj * LDT + 16 + khalf];
    }
#pragma unroll
    for (int j = 0; j < 3; j++)
      acc[j] = __builtin_amdgcn_wmma_f32_16x16x32_bf16(
          false, aF.f, false, bF[j].f, (short)0, acc[j], false, false);
    __syncthreads();
  }

  int col = lane & 15;
#pragma unroll
  for (int j = 0; j < 3; j++)
#pragma unroll
    for (int v = 0; v < 8; v++) {
      int row = (lane < 16) ? v : (8 + v);
      int p2 = bm + wv * 16 + row;
      int n = j * 16 + col;
      if (p2 < NPOS && n < 45) {
        float x = acc[j][v];
        if (n < 9) logits[p2 * 9 + n] = x + cls_b[n];
        else box_pred[p2 * 36 + (n - 9)] = x + bbox_b[n - 9];
      }
    }
}

// ---------------- anchors + decode + sigmoid ----------------
__global__ void k_decode(const float* __restrict__ logits, const float* __restrict__ box_pred,
                         float* __restrict__ anchors, float* __restrict__ proposals,
                         float* __restrict__ scores) {
  int i = blockIdx.x * 256 + threadIdx.x;
  if (i >= NANCH) return;
  int a = i % 9, pos = i / 9;
  int x = pos % GW, y = pos / GW;
  const float scl[3] = {64.f, 128.f, 256.f};
  const float rat[3] = {0.5f, 1.0f, 1.5f};
  int ri = a / 3, si = a - ri * 3;
  float hr = sqrtf(rat[ri]);
  float wsv = scl[si] / hr;
  float hsv = scl[si] * hr;
  float bx0 = rintf(-wsv * 0.5f), by0 = rintf(-hsv * 0.5f);
  float bx1 = rintf(wsv * 0.5f), by1 = rintf(hsv * 0.5f);
  float ax0 = x * 16.f + bx0, ay0 = y * 16.f + by0;
  float ax1 = x * 16.f + bx1, ay1 = y * 16.f + by1;
  anchors[i * 4 + 0] = ax0; anchors[i * 4 + 1] = ay0;
  anchors[i * 4 + 2] = ax1; anchors[i * 4 + 3] = ay1;
  float aw = ax1 - ax0, ah = ay1 - ay0;
  float acx = ax0 + 0.5f * aw, acy = ay0 + 0.5f * ah;
  float dx = box_pred[i * 4 + 0], dy = box_pred[i * 4 + 1];
  float dw = box_pred[i * 4 + 2], dh = box_pred[i * 4 + 3];
  float pcx = dx * aw + acx, pcy = dy * ah + acy;
  float pw = expf(dw) * aw, ph = expf(dh) * ah;
  proposals[i * 4 + 0] = pcx - 0.5f * pw;
  proposals[i * 4 + 1] = pcy - 0.5f * ph;
  proposals[i * 4 + 2] = pcx + 0.5f * pw;
  proposals[i * 4 + 3] = pcy + 0.5f * ph;
  scores[i] = 1.f / (1.f + expf(-logits[i]));
}

// ---------------- sort keys ----------------
__global__ void k_fillkeys(const float* __restrict__ scores, unsigned long long* __restrict__ keys) {
  int i = blockIdx.x * 256 + threadIdx.x;
  if (i >= NKEYS) return;
  if (i < NANCH) {
    unsigned u = __float_as_uint(scores[i]);
    unsigned m = (u & 0x80000000u) ? ~u : (u | 0x80000000u);
    keys[i] = ((unsigned long long)(~m) << 32) | (unsigned)i;
  } else {
    keys[i] = 0xFFFFFFFFFFFFFFFFull;
  }
}

__global__ void k_bitonic(unsigned long long* keys, unsigned j, unsigned k) {
  unsigned i = blockIdx.x * 256 + threadIdx.x;
  unsigned ixj = i ^ j;
  if (ixj > i) {
    unsigned long long a = keys[i], b = keys[ixj];
    bool up = ((i & k) == 0);
    bool swap = up ? (a > b) : (a < b);
    if (swap) { keys[i] = b; keys[ixj] = a; }
  }
}

// ---------------- gather + clip + min-size ----------------
__global__ void k_gather(const unsigned long long* __restrict__ keys,
                         const float* __restrict__ proposals, const float* __restrict__ scores,
                         float* __restrict__ pr, float* __restrict__ sc, int* __restrict__ valid) {
  int t = blockIdx.x * 256 + threadIdx.x;
  if (t >= NTOP) return;
  unsigned idx = (unsigned)(keys[t] & 0xFFFFFFFFull);
  float x0 = fminf(fmaxf(proposals[idx * 4 + 0], 0.f), IMG);
  float y0 = fminf(fmaxf(proposals[idx * 4 + 1], 0.f), IMG);
  float x1 = fminf(fmaxf(proposals[idx * 4 + 2], 0.f), IMG);
  float y1 = fminf(fmaxf(proposals[idx * 4 + 3], 0.f), IMG);
  pr[t * 4 + 0] = x0; pr[t * 4 + 1] = y0; pr[t * 4 + 2] = x1; pr[t * 4 + 3] = y1;
  sc[t] = scores[idx];
  valid[t] = ((x1 - x0) >= 16.f && (y1 - y0) >= 16.f) ? 1 : 0;
}

// ---------------- NMS bitmask ----------------
__global__ void k_nmsmask(const float* __restrict__ boxes, unsigned long long* __restrict__ mask) {
  int rowB = blockIdx.y, colB = blockIdx.x;
  __shared__ float4 cb[64];
  int colStart = colB * 64;
  int colSize = min(64, NTOP - colStart);
  if ((int)threadIdx.x < colSize) cb[threadIdx.x] = ((const float4*)boxes)[colStart + threadIdx.x];
  __syncthreads();
  int i = rowB * 64 + threadIdx.x;
  if (i >= NTOP) return;
  unsigned long long bits = 0ull;
  if (colB >= rowB) {
    float4 rb = ((const float4*)boxes)[i];
    float areaI = (rb.z - rb.x) * (rb.w - rb.y);
    int start = (rowB == colB) ? ((int)threadIdx.x + 1) : 0;
    for (int j = start; j < colSize; j++) {
      float4 b = cb[j];
      float lx = fmaxf(rb.x, b.x), ly = fmaxf(rb.y, b.y);
      float rx = fminf(rb.z, b.z), ry = fminf(rb.w, b.w);
      float w = fmaxf(rx - lx, 0.f), h = fmaxf(ry - ly, 0.f);
      float inter = w * h;
      float areaJ = (b.z - b.x) * (b.w - b.y);
      float iou = inter / (areaI + areaJ - inter + 1e-9f);
      if (iou > 0.7f) bits |= (1ull << j);
    }
  }
  mask[(size_t)i * NWORDS + colB] = bits;
}

__global__ void k_nmsreduce(const unsigned long long* __restrict__ mask,
                            const int* __restrict__ valid, int* __restrict__ keep) {
  __shared__ unsigned long long remv[NWORDS];
  __shared__ int kflag;
  int t = threadIdx.x;
  for (int w = t; w < NWORDS; w += blockDim.x) remv[w] = 0ull;
  __syncthreads();
  for (int i = 0; i < NTOP; i++) {
    if (t == 0) {
      int w = i >> 6, b = i & 63;
      int k = (valid[i] && !((remv[w] >> b) & 1ull)) ? 1 : 0;
      keep[i] = k;
      kflag = k;
    }
    __syncthreads();
    if (kflag) {
      for (int w = t; w < NWORDS; w += blockDim.x) remv[w] |= mask[(size_t)i * NWORDS + w];
    }
    __syncthreads();
  }
}

__global__ void k_compact(const int* __restrict__ keep, const float* __restrict__ boxes,
                          const float* __restrict__ sc, float* __restrict__ out) {
  if (blockIdx.x == 0 && threadIdx.x == 0) {
    int cnt = 0;
    for (int i = 0; i < NTOP && cnt < NOUT; i++) {
      if (keep[i]) {
        out[cnt * 4 + 0] = boxes[i * 4 + 0];
        out[cnt * 4 + 1] = boxes[i * 4 + 1];
        out[cnt * 4 + 2] = boxes[i * 4 + 2];
        out[cnt * 4 + 3] = boxes[i * 4 + 3];
        out[8000 + cnt] = sc[i];
        cnt++;
      }
    }
  }
}

// ---------------- best IoU per gt ----------------
__global__ void k_bpg(const float* __restrict__ gt, const float* __restrict__ anchors,
                      float* __restrict__ bpg) {
  __shared__ float red[256];
  int g = blockIdx.x;
  float gx0 = gt[g * 4 + 0], gy0 = gt[g * 4 + 1], gx1 = gt[g * 4 + 2], gy1 = gt[g * 4 + 3];
  float m = -1e30f;
  for (int i = threadIdx.x; i < NANCH; i += 256) {
    float v = iou_ga(gx0, gy0, gx1, gy1,
                     anchors[i * 4 + 0], anchors[i * 4 + 1], anchors[i * 4 + 2], anchors[i * 4 + 3]);
    m = fmaxf(m, v);
  }
  red[threadIdx.x] = m;
  __syncthreads();
  for (int s = 128; s > 0; s >>= 1) {
    if ((int)threadIdx.x < s) red[threadIdx.x] = fmaxf(red[threadIdx.x], red[threadIdx.x + s]);
    __syncthreads();
  }
  if (threadIdx.x == 0) bpg[g] = red[0];
}

// ---------------- per-anchor targets ----------------
__global__ void k_targets(const float* __restrict__ gt, const float* __restrict__ anchors,
                          const float* __restrict__ bpg, float* __restrict__ labels,
                          float* __restrict__ regt, int* __restrict__ blockPos,
                          int* __restrict__ blockNeg) {
  __shared__ float sg[64];
  __shared__ float sb[16];
  __shared__ int cpos, cneg;
  int t = threadIdx.x;
  if (t < 64) sg[t] = gt[t];
  if (t < 16) sb[t] = bpg[t];
  if (t == 0) { cpos = 0; cneg = 0; }
  __syncthreads();
  int i = blockIdx.x * 256 + t;
  if (i < NANCH) {
    float ax0 = anchors[i * 4 + 0], ay0 = anchors[i * 4 + 1];
    float ax1 = anchors[i * 4 + 2], ay1 = anchors[i * 4 + 3];
    float best = -1e30f;
    int pre = 0, force = 0;
    for (int g = 0; g < 16; g++) {
      float v = iou_ga(sg[g * 4 + 0], sg[g * 4 + 1], sg[g * 4 + 2], sg[g * 4 + 3],
                       ax0, ay0, ax1, ay1);
      if (v > best) { best = v; pre = g; }
      if (v == sb[g]) force = 1;
    }
    int midx = (best < 0.3f) ? -1 : ((best < 0.7f) ? -2 : pre);
    if (force) midx = pre;
    float lab = (midx >= 0) ? 1.f : ((midx == -2) ? -1.f : 0.f);
    labels[i] = lab;
    int mg = (midx < 0) ? 0 : midx;
    float mx0 = sg[mg * 4 + 0], my0 = sg[mg * 4 + 1], mx1 = sg[mg * 4 + 2], my1 = sg[mg * 4 + 3];
    float aw = ax1 - ax0, ah = ay1 - ay0;
    float acx = ax0 + 0.5f * aw, acy = ay0 + 0.5f * ah;
    float gw_ = mx1 - mx0, gh_ = my1 - my0;
    float gcx = mx0 + 0.5f * gw_, gcy = my0 + 0.5f * gh_;
    regt[i * 4 + 0] = (gcx - acx) / aw;
    regt[i * 4 + 1] = (gcy - acy) / ah;
    regt[i * 4 + 2] = logf(gw_ / aw);
    regt[i * 4 + 3] = logf(gh_ / ah);
    if (lab == 1.f) atomicAdd(&cpos, 1);
    if (lab == 0.f) atomicAdd(&cneg, 1);
  }
  __syncthreads();
  if (t == 0) { blockPos[blockIdx.x] = cpos; blockNeg[blockIdx.x] = cneg; }
}

// ---------------- scan of block counts ----------------
__global__ void k_scan(const int* __restrict__ bp, const int* __restrict__ bn,
                       int* __restrict__ posOff, int* __restrict__ negOff,
                       int* __restrict__ sci, float* __restrict__ scf, float* __restrict__ accs) {
  __shared__ int sp[512], sn[512];
  int t = threadIdx.x;
  sp[t] = (t < NBLK) ? bp[t] : 0;
  sn[t] = (t < NBLK) ? bn[t] : 0;
  __syncthreads();
  for (int d = 1; d < 512; d <<= 1) {
    int vp = (t >= d) ? sp[t - d] : 0;
    int vn = (t >= d) ? sn[t - d] : 0;
    __syncthreads();
    sp[t] += vp; sn[t] += vn;
    __syncthreads();
  }
  if (t < NBLK) {
    posOff[t] = sp[t] - bp[t];
    negOff[t] = sn[t] - bn[t];
  }
  if (t == 0) {
    int totPos = sp[511], totNeg = sn[511];
    int nPos = min(totPos, 128);
    int T = 256 - nPos;
    int nSampNeg = min(totNeg, T);
    sci[0] = T;
    scf[0] = (float)(nPos + nSampNeg);
    accs[0] = 0.f; accs[1] = 0.f;
  }
}

// ---------------- losses ----------------
__global__ void k_loss(const float* __restrict__ labels, const float* __restrict__ box_pred,
                       const float* __restrict__ regt, const float* __restrict__ logits,
                       const int* __restrict__ posOff, const int* __restrict__ negOff,
                       const int* __restrict__ sci, float* __restrict__ accs) {
  __shared__ int sp[256], sn[256];
  __shared__ float red[256];
  int t = threadIdx.x;
  int i = blockIdx.x * 256 + t;
  float lab = (i < NANCH) ? labels[i] : -1.f;
  int isPos = (lab == 1.f) ? 1 : 0;
  int isNeg = (lab == 0.f) ? 1 : 0;
  sp[t] = isPos; sn[t] = isNeg;
  __syncthreads();
  for (int d = 1; d < 256; d <<= 1) {
    int vp = (t >= d) ? sp[t - d] : 0;
    int vn = (t >= d) ? sn[t - d] : 0;
    __syncthreads();
    sp[t] += vp; sn[t] += vn;
    __syncthreads();
  }
  int T = sci[0];
  int cumPos = posOff[blockIdx.x] + sp[t];
  int cumNeg = negOff[blockIdx.x] + sn[t];
  int pos_sel = isPos && (cumPos <= 128);
  int neg_sel = isNeg && (cumNeg <= T);
  int samp = pos_sel | neg_sel;
  float locSum = 0.f, clsSum = 0.f;
  if (i < NANCH) {
    if (pos_sel) {
      const float beta = 1.f / 9.f;
#pragma unroll
      for (int j = 0; j < 4; j++) {
        float d = box_pred[i * 4 + j] - regt[i * 4 + j];
        float ad = fabsf(d);
        locSum += (ad < beta) ? (0.5f * d * d / beta) : (ad - 0.5f * beta);
      }
    }
    if (samp) {
      float x = logits[i], y = lab;
      clsSum = fmaxf(x, 0.f) - x * y + log1pf(expf(-fabsf(x)));
    }
  }
  red[t] = clsSum;
  __syncthreads();
  for (int s = 128; s > 0; s >>= 1) {
    if (t < s) red[t] += red[t + s];
    __syncthreads();
  }
  if (t == 0) atomicAdd(&accs[0], red[0]);
  __syncthreads();
  red[t] = locSum;
  __syncthreads();
  for (int s = 128; s > 0; s >>= 1) {
    if (t < s) red[t] += red[t + s];
    __syncthreads();
  }
  if (t == 0) atomicAdd(&accs[1], red[0]);
}

__global__ void k_fin(const float* __restrict__ accs, const float* __restrict__ scf,
                      float* __restrict__ out) {
  if (blockIdx.x == 0 && threadIdx.x == 0) {
    float ns = scf[0];
    out[10000] = accs[0] / ns;  // cls_loss
    out[10001] = accs[1] / ns;  // loc_loss
  }
}

extern "C" void kernel_launch(void* const* d_in, const int* in_sizes, int n_in,
                              void* d_out, int out_size, void* d_ws, size_t ws_size,
                              hipStream_t stream) {
  (void)in_sizes; (void)n_in; (void)out_size; (void)ws_size;
  const float* feat   = (const float*)d_in[1];
  const float* gt     = (const float*)d_in[2];
  const float* rpn_w  = (const float*)d_in[3];
  const float* rpn_b  = (const float*)d_in[4];
  const float* cls_w  = (const float*)d_in[5];
  const float* cls_b  = (const float*)d_in[6];
  const float* bbox_w = (const float*)d_in[7];
  const float* bbox_b = (const float*)d_in[8];
  float* out = (float*)d_out;

  char* p = (char*)d_ws;
  auto alloc = [&](size_t bytes) -> void* {
    void* r = (void*)p;
    p += (bytes + 255) & ~(size_t)255;
    return r;
  };
  unsigned short* fp   = (unsigned short*)alloc((size_t)102 * 102 * 512 * 2);  // dead after gemm1
  unsigned short* bw   = (unsigned short*)alloc((size_t)512 * KTOT * 2);       // dead after gemm1
  unsigned short* rf   = (unsigned short*)alloc((size_t)NPOS * 512 * 2);
  unsigned short* b2   = (unsigned short*)alloc((size_t)48 * 512 * 2);
  float* logits        = (float*)alloc((size_t)NANCH * 4);
  float* box_pred      = (float*)alloc((size_t)NANCH * 4 * 4);
  float* anchors       = (float*)alloc((size_t)NANCH * 4 * 4);
  float* proposals     = (float*)alloc((size_t)NANCH * 4 * 4);
  float* scores        = (float*)alloc((size_t)NANCH * 4);
  unsigned long long* keys = (unsigned long long*)alloc((size_t)NKEYS * 8);
  float* pr            = (float*)alloc((size_t)NTOP * 4 * 4);
  float* sc            = (float*)alloc((size_t)NTOP * 4);
  int* valid           = (int*)alloc((size_t)NTOP * 4);
  int* keep            = (int*)alloc((size_t)NTOP * 4);
  float* bpg           = (float*)alloc(16 * 4);
  float* labels        = (float*)alloc((size_t)NANCH * 4);
  float* regt          = (float*)alloc((size_t)NANCH * 4 * 4);
  int* blockPos        = (int*)alloc(NBLK * 4);
  int* blockNeg        = (int*)alloc(NBLK * 4);
  int* posOff          = (int*)alloc(NBLK * 4);
  int* negOff          = (int*)alloc(NBLK * 4);
  int* sci             = (int*)alloc(8 * 4);
  float* scf           = (float*)alloc(8 * 4);
  float* accs          = (float*)alloc(8 * 4);
  // NMS masks alias the fp+bw region (both dead once gemm1 completes).
  unsigned long long* mask = (unsigned long long*)fp;  // 10000*157*8 = 12.56MB <= 15.3MB

  k_init_out<<<(10002 + 255) / 256, 256, 0, stream>>>(out, 10002);
  k_featpad<<<(102 * 102 * 512 + 255) / 256, 256, 0, stream>>>(feat, fp);
  k_wconv<<<(512 * KTOT + 255) / 256, 256, 0, stream>>>(rpn_w, bw);
  k_w2<<<(48 * 512 + 255) / 256, 256, 0, stream>>>(cls_w, bbox_w, b2);
  k_gemm1<<<dim3(157, 4), 256, 0, stream>>>(fp, bw, rpn_b, rf);
  k_gemm2<<<79, 256, 0, stream>>>(rf, b2, cls_b, bbox_b, logits, box_pred);
  k_decode<<<NBLK, 256, 0, stream>>>(logits, box_pred, anchors, proposals, scores);
  k_fillkeys<<<NKEYS / 256, 256, 0, stream>>>(scores, keys);
  for (unsigned kk = 2; kk <= NKEYS; kk <<= 1)
    for (unsigned jj = kk >> 1; jj >= 1; jj >>= 1)
      k_bitonic<<<NKEYS / 256, 256, 0, stream>>>(keys, jj, kk);
  k_gather<<<(NTOP + 255) / 256, 256, 0, stream>>>(keys, proposals, scores, pr, sc, valid);
  k_nmsmask<<<dim3(NWORDS, NWORDS), 64, 0, stream>>>(pr, mask);
  k_nmsreduce<<<1, 256, 0, stream>>>(mask, valid, keep);
  k_compact<<<1, 32, 0, stream>>>(keep, pr, sc, out);
  k_bpg<<<16, 256, 0, stream>>>(gt, anchors, bpg);
  k_targets<<<NBLK, 256, 0, stream>>>(gt, anchors, bpg, labels, regt, blockPos, blockNeg);
  k_scan<<<1, 512, 0, stream>>>(blockPos, blockNeg, posOff, negOff, sci, scf, accs);
  k_loss<<<NBLK, 256, 0, stream>>>(labels, box_pred, regt, logits, posOff, negOff, sci, accs);
  k_fin<<<1, 32, 0, stream>>>(accs, scf, out);
}
